// GCN_29686813950829
// MI455X (gfx1250) — matrix operand order repaired and checked by reading, not compiled, not run on hardware
//
#include <hip/hip_runtime.h>
#include <hip/hip_bf16.h>
#include <math.h>

// ---------------------------------------------------------------------------
// GCN forward on MI455X (gfx1250): bf16 WMMA GEMMs + atomic scatter aggregation
// ---------------------------------------------------------------------------

#define F_IN   1433
#define H_DIM  256
#define C_DIM  7

typedef __bf16 bf16x16 __attribute__((ext_vector_type(16)));
typedef float  f32x8   __attribute__((ext_vector_type(8)));

union FragAB { unsigned u[8]; bf16x16 v; };

__device__ __forceinline__ unsigned short f2bf_raw(float f) {
    // round-to-nearest-even f32 -> bf16 bit pattern
    unsigned u = __builtin_bit_cast(unsigned, f);
    unsigned r = u + 0x7FFFu + ((u >> 16) & 1u);
    return (unsigned short)(r >> 16);
}

__device__ __forceinline__ unsigned pack_bf16(float lo, float hi) {
    return (unsigned)f2bf_raw(lo) | ((unsigned)f2bf_raw(hi) << 16);
}

__device__ __forceinline__ int imin(int a, int b) { return a < b ? a : b; }

// ------------------------- degree / normalization --------------------------

__global__ void k_fill1(float* __restrict__ p, int n) {
    int i = blockIdx.x * blockDim.x + threadIdx.x;
    if (i < n) p[i] = 1.0f;                       // self-loop contribution
}

__global__ void k_deg(const int* __restrict__ col, float* __restrict__ deg, int E) {
    int e = blockIdx.x * blockDim.x + threadIdx.x;
    if (e < E) atomicAdd(&deg[col[e]], 1.0f);
}

__global__ void k_rsqrt(float* __restrict__ p, int n) {
    int i = blockIdx.x * blockDim.x + threadIdx.x;
    if (i < n) p[i] = rsqrtf(p[i]);               // deg >= 1 always (self-loops)
}

// ----------------------------- WMMA GEMM -----------------------------------
// C[M,Nc] = A[M,K] @ W[K,Nc]; f32 in/out, bf16 WMMA compute, f32 accumulate.
// Block tile 128x64, BK=32. 8 waves (wave32): 4 (M) x 2 (N), each wave 32x32.
// K-loop is peeled: kFull unconditional tiles + one clamped tail tile.

#define BM 128
#define BN 64
#define BK 32
#define LDT (BK + 4)   // LDS row pitch in shorts (pad to dodge bank conflicts)

__device__ __forceinline__ void mma_step(
    const unsigned short (&As)[BM][LDT], const unsigned short (&Bs)[BN][LDT],
    f32x8 (&acc)[2][2], int wm0, int wn0, int mrow, int halfs)
{
    // A 16x32: lane L holds row (L%16); VGPR j<4: K = (L>=16?8:0)+2j,+1
    //                                   VGPR j>=4: K = 16+(L>=16?8:0)+2(j-4),+1
    FragAB afrag[2], bfrag[2];
    #pragma unroll
    for (int tm = 0; tm < 2; ++tm) {
        const unsigned* arow = (const unsigned*)&As[wm0 + tm * 16 + mrow][0];
        const int baseu = halfs * 4;
        #pragma unroll
        for (int j = 0; j < 4; ++j) {
            afrag[tm].u[j]     = arow[baseu + j];
            afrag[tm].u[4 + j] = arow[8 + baseu + j];
        }
    }
    // B 32x16: lane L holds col (L%16); same K layout as A (j<4: (L>=16?8:0)+2j; j>=4: 16+(L>=16?8:0)+2(j-4))
    #pragma unroll
    for (int tn = 0; tn < 2; ++tn) {
        const unsigned* brow = (const unsigned*)&Bs[wn0 + tn * 16 + mrow][0];
        const int baseu = halfs * 4;
        #pragma unroll
        for (int j = 0; j < 8; ++j) bfrag[tn].u[j] = brow[(j >> 2) * 8 + baseu + (j & 3)];
    }
    #pragma unroll
    for (int tm = 0; tm < 2; ++tm)
        #pragma unroll
        for (int tn = 0; tn < 2; ++tn)
            acc[tm][tn] = __builtin_amdgcn_wmma_f32_16x16x32_bf16(
                false, afrag[tm].v, false, bfrag[tn].v,
                (short)0, acc[tm][tn], false, false);
}

template <bool VECA>
__global__ __launch_bounds__(256) void k_gemm_wmma(
    const float* __restrict__ A, const float* __restrict__ W,
    float* __restrict__ C, int M, int K, int Nc)
{
    __shared__ unsigned short As[BM][LDT];   // As[m][k]  (k pairs contiguous)
    __shared__ unsigned short Bs[BN][LDT];   // Bs[n][k]  (transposed: k pairs contiguous)

    const int tid  = threadIdx.x;
    const int lane = tid & 31;
    const int wave = tid >> 5;
    const int wm0  = (wave >> 1) * 32;       // wave's M offset within block tile
    const int wn0  = (wave & 1)  * 32;       // wave's N offset within block tile
    const int row0 = blockIdx.x * BM;
    const int col0 = blockIdx.y * BN;

    const f32x8 zero = {0.f,0.f,0.f,0.f,0.f,0.f,0.f,0.f};
    f32x8 acc[2][2] = {{zero, zero}, {zero, zero}};

    const int mrow  = lane & 15;             // row within 16x16 tile (A) / col (B,C)
    const int halfs = lane >> 4;             // lane half selects K sub-range

    // staging coordinates (loop-invariant). Rows are clamped, never predicated:
    // garbage in M-tail rows only affects discarded output rows.
    const int a_m   = (tid * 16) >> 5;       // 0..127
    const int a_kk  = (tid * 16) & 31;       // 0 or 16
    const float* rowA = A + (long)imin(row0 + a_m, M - 1) * K;  // row base
    const float* srcA = rowA + a_kk;
    unsigned* dstA  = (unsigned*)&As[a_m][a_kk];    // 4B aligned, b64-mergeable
    const int b_kk  = (tid * 8) >> 6;        // 0..31
    const int b_nn  = (tid * 8) & 63;
    const float* srcB = W + (long)b_kk * Nc + col0 + b_nn;   // 32B aligned

    const int kFull = K / BK;                // fully-populated K tiles
    for (int kt = 0; kt < kFull; ++kt) {
        const int k0 = kt * BK;
        // ---- stage A tile (128x32): unconditional loads ----
        {
            const float* s = srcA + k0;
            __builtin_prefetch(s + BK, 0, 3);
            unsigned pk[8];
            if (VECA) {
                float4 f[4];
                #pragma unroll
                for (int q = 0; q < 4; ++q) f[q] = *(const float4*)(s + 4 * q);
                const float* ff = (const float*)f;
                #pragma unroll
                for (int p = 0; p < 8; ++p) pk[p] = pack_bf16(ff[2 * p], ff[2 * p + 1]);
            } else {
                #pragma unroll
                for (int p = 0; p < 8; ++p) pk[p] = pack_bf16(s[2 * p], s[2 * p + 1]);
            }
            #pragma unroll
            for (int p = 0; p < 8; ++p) dstA[p] = pk[p];
        }
        // ---- stage B tile (32x64) transposed: unconditional float4 loads ----
        {
            const float* s = srcB + (long)k0 * Nc;
            const float4 f0 = *(const float4*)s;
            const float4 f1 = *(const float4*)(s + 4);
            const float ff[8] = {f0.x,f0.y,f0.z,f0.w,f1.x,f1.y,f1.z,f1.w};
            #pragma unroll
            for (int i = 0; i < 8; ++i) Bs[b_nn + i][b_kk] = f2bf_raw(ff[i]);
        }
        __syncthreads();
        mma_step(As, Bs, acc, wm0, wn0, mrow, halfs);
        __syncthreads();
    }

    // ---- peeled partial tail tile (layer 1 only: K % 32 != 0) ----
    if (kFull * BK < K) {
        const int k0 = kFull * BK;
        {   // A: clamped address off the row base, value select to zero
            unsigned pk[8];
            #pragma unroll
            for (int p = 0; p < 8; ++p) {
                const int kg0 = k0 + a_kk + 2 * p;   // true global k
                const int kg1 = kg0 + 1;
                float f0 = rowA[imin(kg0, K - 1)];
                float f1 = rowA[imin(kg1, K - 1)];
                f0 = (kg0 < K) ? f0 : 0.0f;
                f1 = (kg1 < K) ? f1 : 0.0f;
                pk[p] = pack_bf16(f0, f1);
            }
            #pragma unroll
            for (int p = 0; p < 8; ++p) dstA[p] = pk[p];
        }
        {   // B: clamped row (relative to srcB's built-in b_kk), select to zero
            const int kg = k0 + b_kk;                // true global k of this row
            const float* s = srcB + (long)(imin(kg, K - 1) - b_kk) * Nc;
            float4 f0 = *(const float4*)s;
            float4 f1 = *(const float4*)(s + 4);
            const bool kok = (kg < K);
            float ff[8] = {f0.x,f0.y,f0.z,f0.w,f1.x,f1.y,f1.z,f1.w};
            #pragma unroll
            for (int i = 0; i < 8; ++i) Bs[b_nn + i][b_kk] = f2bf_raw(kok ? ff[i] : 0.0f);
        }
        __syncthreads();
        mma_step(As, Bs, acc, wm0, wn0, mrow, halfs);
        __syncthreads();
    }

    // ---- write out: C/D layout VGPR r: M = r + (lane>=16 ? 8 : 0), N = lane%16
    #pragma unroll
    for (int tm = 0; tm < 2; ++tm) {
        #pragma unroll
        for (int tn = 0; tn < 2; ++tn) {
            const int n = col0 + wn0 + tn * 16 + mrow;
            #pragma unroll
            for (int r = 0; r < 8; ++r) {
                const int m = row0 + wm0 + tm * 16 + halfs * 8 + r;
                if (m < M) C[(long)m * Nc + n] = acc[tm][tn][r];
            }
        }
    }
}

// --------------------------- graph aggregation -----------------------------

// out[i*F+f] = h[i*F+f] * dinv[i]^2   (self-loop term; also inits the buffer)
__global__ void k_agg_init(const float* __restrict__ h, const float* __restrict__ dinv,
                           float* __restrict__ out, long total, int F) {
    long idx = (long)blockIdx.x * blockDim.x + threadIdx.x;
    if (idx >= total) return;
    int i = (int)(idx / F);
    float d = dinv[i];
    out[idx] = h[idx] * d * d;
}

// F = 256: 64 threads per edge, float4 per thread, 4 f32 atomics.
__global__ __launch_bounds__(256) void k_agg_edges256(
    const float* __restrict__ h, const int* __restrict__ row,
    const int* __restrict__ col, const float* __restrict__ dinv,
    float* __restrict__ out, int E)
{
    long tid = (long)blockIdx.x * blockDim.x + threadIdx.x;
    if (tid >= (long)E * 64) return;
    const int e = (int)(tid >> 6);
    const int q = ((int)tid & 63) << 2;
    const int r = row[e], c = col[e];
    const float nrm = dinv[r] * dinv[c];
    const float4 v = *(const float4*)(h + (long)r * 256 + q);
    float* o = out + (long)c * 256 + q;
    atomicAdd(o + 0, v.x * nrm);
    atomicAdd(o + 1, v.y * nrm);
    atomicAdd(o + 2, v.z * nrm);
    atomicAdd(o + 3, v.w * nrm);
}

// F = 7: one thread per edge.
__global__ void k_agg_edges7(const float* __restrict__ h, const int* __restrict__ row,
                             const int* __restrict__ col, const float* __restrict__ dinv,
                             float* __restrict__ out, int E)
{
    int e = blockIdx.x * blockDim.x + threadIdx.x;
    if (e >= E) return;
    const int r = row[e], c = col[e];
    const float nrm = dinv[r] * dinv[c];
    const float* hv = h + (long)r * C_DIM;
    float* o = out + (long)c * C_DIM;
    #pragma unroll
    for (int j = 0; j < C_DIM; ++j) atomicAdd(o + j, hv[j] * nrm);
}

__global__ void k_bias_relu(float* __restrict__ a, const float* __restrict__ b, long total) {
    long idx = (long)blockIdx.x * blockDim.x + threadIdx.x;
    if (idx >= total) return;
    const int f = (int)(idx & (H_DIM - 1));
    a[idx] = fmaxf(a[idx] + b[f], 0.0f);
}

// --------------------- layer 3: [N,256] @ [256,7] --------------------------
// one wave per node; W3 cached in LDS; shuffle reduction across 32 lanes.
__global__ __launch_bounds__(256) void k_gemm_h7(
    const float* __restrict__ A, const float* __restrict__ W,
    float* __restrict__ out, int M)
{
    __shared__ float Ws[H_DIM * C_DIM];
    for (int i = threadIdx.x; i < H_DIM * C_DIM; i += 256) Ws[i] = W[i];
    __syncthreads();

    const int lane = threadIdx.x & 31;
    const int wave = threadIdx.x >> 5;
    const int node = blockIdx.x * 8 + wave;
    if (node >= M) return;

    float acc[C_DIM] = {0.f, 0.f, 0.f, 0.f, 0.f, 0.f, 0.f};
    const float* arow = A + (long)node * H_DIM;
    #pragma unroll
    for (int s = 0; s < 8; ++s) {
        const int k = s * 32 + lane;
        const float av = arow[k];
        const float* w = &Ws[k * C_DIM];
        #pragma unroll
        for (int j = 0; j < C_DIM; ++j) acc[j] += av * w[j];
    }
    #pragma unroll
    for (int off = 16; off >= 1; off >>= 1)
        #pragma unroll
        for (int j = 0; j < C_DIM; ++j)
            acc[j] += __shfl_down(acc[j], off, 32);
    if (lane == 0) {
        float* o = out + (long)node * C_DIM;
        #pragma unroll
        for (int j = 0; j < C_DIM; ++j) o[j] = acc[j];
    }
}

__global__ void k_bias_logsoftmax(const float* __restrict__ g, const float* __restrict__ b,
                                  float* __restrict__ out, int n)
{
    int i = blockIdx.x * blockDim.x + threadIdx.x;
    if (i >= n) return;
    float v[C_DIM];
    float mx = -3.4e38f;
    #pragma unroll
    for (int j = 0; j < C_DIM; ++j) {
        v[j] = g[(long)i * C_DIM + j] + b[j];
        mx = fmaxf(mx, v[j]);
    }
    float s = 0.0f;
    #pragma unroll
    for (int j = 0; j < C_DIM; ++j) s += expf(v[j] - mx);
    const float l = logf(s);
    #pragma unroll
    for (int j = 0; j < C_DIM; ++j) out[(long)i * C_DIM + j] = v[j] - mx - l;
}

// ------------------------------- launcher ----------------------------------

extern "C" void kernel_launch(void* const* d_in, const int* in_sizes, int n_in,
                              void* d_out, int out_size, void* d_ws, size_t ws_size,
                              hipStream_t stream)
{
    const float* x  = (const float*)d_in[0];
    const int*   ei = (const int*)  d_in[1];
    const float* W1 = (const float*)d_in[2];
    const float* b1 = (const float*)d_in[3];
    const float* W2 = (const float*)d_in[4];
    const float* b2 = (const float*)d_in[5];
    const float* W3 = (const float*)d_in[6];
    const float* b3 = (const float*)d_in[7];

    const int N = in_sizes[0] / F_IN;      // 50000
    const int E = in_sizes[1] / 2;         // 800000
    const int* row = ei;
    const int* col = ei + E;

    // workspace carve (256B aligned)
    char* wsb = (char*)d_ws;
    size_t off = 0;
    auto carve = [&](size_t bytes) {
        void* p = wsb + off;
        off = (off + bytes + 255) & ~(size_t)255;
        return p;
    };
    float* dinv = (float*)carve((size_t)N * 4);               // deg -> dinv in place
    float* h    = (float*)carve((size_t)N * H_DIM * 4);       // GEMM output
    float* a    = (float*)carve((size_t)N * H_DIM * 4);       // aggregated/activated
    float* h3   = (float*)carve((size_t)N * C_DIM * 4);
    float* g3   = (float*)carve((size_t)N * C_DIM * 4);
    (void)ws_size;

    const int T = 256;
    const long NH = (long)N * H_DIM;
    const long NC = (long)N * C_DIM;

    // normalization: deg = 1 + in-degree; dinv = rsqrt(deg)
    k_fill1<<<(N + T - 1) / T, T, 0, stream>>>(dinv, N);
    k_deg  <<<(E + T - 1) / T, T, 0, stream>>>(col, dinv, E);
    k_rsqrt<<<(N + T - 1) / T, T, 0, stream>>>(dinv, N);

    dim3 gemmGrid((N + BM - 1) / BM, H_DIM / BN);

    // ---- layer 1 ----  (K = 1433: rows only 4B-aligned -> scalar A loads)
    k_gemm_wmma<false><<<gemmGrid, T, 0, stream>>>(x, W1, h, N, F_IN, H_DIM);
    k_agg_init<<<(NH + T - 1) / T, T, 0, stream>>>(h, dinv, a, NH, H_DIM);
    k_agg_edges256<<<(int)(((long)E * 64 + T - 1) / T), T, 0, stream>>>(h, row, col, dinv, a, E);
    k_bias_relu<<<(NH + T - 1) / T, T, 0, stream>>>(a, b1, NH);

    // ---- layer 2 ----  (K = 256: 16B-aligned rows -> float4 A loads)
    k_gemm_wmma<true><<<gemmGrid, T, 0, stream>>>(a, W2, h, N, H_DIM, H_DIM);
    k_agg_init<<<(NH + T - 1) / T, T, 0, stream>>>(h, dinv, a, NH, H_DIM);
    k_agg_edges256<<<(int)(((long)E * 64 + T - 1) / T), T, 0, stream>>>(h, row, col, dinv, a, E);
    k_bias_relu<<<(NH + T - 1) / T, T, 0, stream>>>(a, b2, NH);

    // ---- layer 3 ----
    k_gemm_h7<<<(N + 7) / 8, T, 0, stream>>>(a, W3, h3, N);
    k_agg_init<<<(NC + T - 1) / T, T, 0, stream>>>(h3, dinv, g3, NC, C_DIM);
    k_agg_edges7<<<(E + T - 1) / T, T, 0, stream>>>(h3, row, col, dinv, g3, E);
    k_bias_logsoftmax<<<(N + T - 1) / T, T, 0, stream>>>(g3, b3, (float*)d_out, N);
}
